// UniMP_23888608101048
// MI455X (gfx1250) — matrix operand order repaired and checked by reading, not compiled, not run on hardware
//
#include <hip/hip_runtime.h>
#include <hip/hip_bf16.h>
#include <math.h>

// ---------------------------------------------------------------------------
// Types (own ext_vector types so they are trivially-copyable / union-safe)
// ---------------------------------------------------------------------------
typedef __bf16 bf16;
typedef __attribute__((ext_vector_type(16))) __bf16  v16bf;
typedef __attribute__((ext_vector_type(8)))  float   v8f;
typedef __attribute__((ext_vector_type(4)))  unsigned int u32x4;
typedef __attribute__((ext_vector_type(4)))  int     i32x4;
typedef __attribute__((ext_vector_type(4)))  float   f32x4;

typedef __attribute__((address_space(1))) i32x4 gv4i;   // global-side v4i32
typedef __attribute__((address_space(3))) i32x4 lv4i;   // LDS-side v4i32
typedef __attribute__((address_space(3))) char  lds_char;

#define NNODES 50000
#define NEDGES 500000
#define HEADS  8

#if defined(__has_builtin)
#if __has_builtin(__builtin_amdgcn_global_load_async_to_lds_b128)
#define HAVE_ASYNC_BUILTIN 1
#endif
#endif

// ---- async global->LDS copy of 16 bytes (one B128 transfer per lane) ------
static __device__ __forceinline__ void async_copy16_g2l(const bf16* gp, bf16* lp) {
#ifdef HAVE_ASYNC_BUILTIN
  __builtin_amdgcn_global_load_async_to_lds_b128((gv4i*)gp, (lv4i*)lp, 0, 0);
#else
  unsigned lofs = (unsigned)(__SIZE_TYPE__)(lds_char*)lp;           // LDS byte offset
  unsigned long long ga = (unsigned long long)(__SIZE_TYPE__)gp;    // 64-bit global addr
  asm volatile("global_load_async_to_lds_b128 %0, %1, off"
               :: "v"(lofs), "v"(ga) : "memory");
#endif
}

static __device__ __forceinline__ void wait_async0() {
#if defined(__has_builtin) && __has_builtin(__builtin_amdgcn_s_wait_asynccnt)
  __builtin_amdgcn_s_wait_asynccnt(0);
#else
  asm volatile("s_wait_asynccnt 0x0" ::: "memory");
#endif
}

static __device__ __forceinline__ float wave_reduce_sum(float v) {
#pragma unroll
  for (int off = 16; off > 0; off >>= 1) v += __shfl_xor(v, off, 32);
  return v;
}

static __device__ __forceinline__ void atomicMaxF(float* addr, float val) {
  // ordering trick valid for all finite floats and +-inf
  if (val >= 0.f) atomicMax(reinterpret_cast<int*>(addr), __float_as_int(val));
  else            atomicMin(reinterpret_cast<unsigned int*>(addr), __float_as_uint(val));
}

// ---------------------------------------------------------------------------
// Elementwise helpers
// ---------------------------------------------------------------------------
__global__ void k_cast_bf16(const float* __restrict__ src, bf16* __restrict__ dst, int n) {
  int i = blockIdx.x * blockDim.x + threadIdx.x;
  if (i < n) dst[i] = (bf16)src[i];
}

__global__ void k_fill(float* __restrict__ p, float v, int n) {
  int i = blockIdx.x * blockDim.x + threadIdx.x;
  if (i < n) p[i] = v;
}

// ---------------------------------------------------------------------------
// WMMA bf16 GEMM:  C[M,Ncol] = A[M,256] @ B[256,Ncol] + bias
// Block tile 128x128, 8 waves, each wave computes 32x64 (2x4 WMMA 16x16 tiles).
// A tile staged to LDS via GLOBAL_LOAD_ASYNC_TO_LDS_B128 (ASYNCcnt path);
// B tile staged transposed [n][k] through VGPRs so fragment loads are
// contiguous ds_load_b128 pairs matching the WMMA A/B VGPR layouts.
// ---------------------------------------------------------------------------
__global__ __launch_bounds__(256)
void k_gemm_bf16(const bf16* __restrict__ A, const bf16* __restrict__ B,
                 const float* __restrict__ bias, float* __restrict__ C,
                 int M, int Ncol) {
  __shared__ __align__(16) bf16 sA[128 * 32];   // [row][k]
  __shared__ __align__(16) bf16 sBT[128 * 32];  // [col][k]

  const int tid  = threadIdx.x;
  const int lane = tid & 31;
  const int wid  = tid >> 5;
  const int m0   = blockIdx.x * 128;
  const int n0   = blockIdx.y * 128;
  const int mw   = (wid & 3) * 32;  // wave row offset inside block tile
  const int nw   = (wid >> 2) * 64; // wave col offset inside block tile

  const v8f vzero = {0.f, 0.f, 0.f, 0.f, 0.f, 0.f, 0.f, 0.f};
  v8f acc[2][4];
#pragma unroll
  for (int i = 0; i < 2; i++)
#pragma unroll
    for (int j = 0; j < 4; j++) acc[i][j] = vzero;

  union FU { v16bf v; u32x4 q[2]; };

  for (int k0 = 0; k0 < 256; k0 += 32) {
    // ---- A tile: 128x32 bf16 = 512 chunks of 8 bf16; 2 async copies/thread ----
#pragma unroll
    for (int j = 0; j < 2; j++) {
      int chunk = tid + j * 256;       // 0..511
      int row   = chunk >> 2;          // 4 chunks per row of 32
      int kc    = (chunk & 3) * 8;
      int gr    = m0 + row;
      if (gr < M) {
        async_copy16_g2l(&A[(size_t)gr * 256 + k0 + kc], &sA[row * 32 + kc]);
      } else {
        u32x4 z = {0u, 0u, 0u, 0u};
        *reinterpret_cast<u32x4*>(&sA[row * 32 + kc]) = z;
      }
    }
    // ---- B tile transposed: read 8 consecutive n per chunk, scatter to LDS ----
#pragma unroll
    for (int j = 0; j < 2; j++) {
      int chunk = tid + j * 256;       // 0..511 ; 16 chunks per k-row
      int kk = chunk >> 4;             // 0..31
      int nc = (chunk & 15) * 8;       // 0..120
      bf16 tmp[8];
      if (n0 + nc + 8 <= Ncol) {
        union { u32x4 q; bf16 h[8]; } u;
        u.q = *reinterpret_cast<const u32x4*>(&B[(size_t)(k0 + kk) * Ncol + n0 + nc]);
#pragma unroll
        for (int t = 0; t < 8; t++) tmp[t] = u.h[t];
      } else {
#pragma unroll
        for (int t = 0; t < 8; t++)
          tmp[t] = (n0 + nc + t < Ncol) ? B[(size_t)(k0 + kk) * Ncol + n0 + nc + t]
                                        : (bf16)0.f;
      }
#pragma unroll
      for (int t = 0; t < 8; t++) sBT[(nc + t) * 32 + kk] = tmp[t];
    }
    wait_async0();          // our async->LDS transfers landed
    __syncthreads();        // everyone's tile writes visible

    // ---- fragments + WMMA ----
    FU afrag[2];
#pragma unroll
    for (int mi = 0; mi < 2; mi++) {
      // A 16x32 bf16 layout: lanes0-15 rows M, K 0..7 & 16..23; lanes16-31 K 8..15 & 24..31
      int row = mw + mi * 16 + (lane & 15);
      int kh  = (lane >> 4) * 8;
      afrag[mi].q[0] = *reinterpret_cast<const u32x4*>(&sA[row * 32 + kh]);
      afrag[mi].q[1] = *reinterpret_cast<const u32x4*>(&sA[row * 32 + 16 + kh]);
    }
#pragma unroll
    for (int ni = 0; ni < 4; ni++) {
      // B 32x16 bf16 layout: lanes0-15 col N K 0..7 & 16..23; lanes16-31 K 8..15 & 24..31
      FU bfrag;
      int col = nw + ni * 16 + (lane & 15);
      int kh  = (lane >> 4) * 8;
      bfrag.q[0] = *reinterpret_cast<const u32x4*>(&sBT[col * 32 + kh]);
      bfrag.q[1] = *reinterpret_cast<const u32x4*>(&sBT[col * 32 + 16 + kh]);
#pragma unroll
      for (int mi = 0; mi < 2; mi++) {
        acc[mi][ni] = __builtin_amdgcn_wmma_f32_16x16x32_bf16(
            false, afrag[mi].v, false, bfrag.v, (short)0, acc[mi][ni], false, false);
      }
    }
    __syncthreads();
  }

  // ---- store: C/D layout: VGPR r -> M = r + 8*(lane>>4), N = lane&15 ----
#pragma unroll
  for (int mi = 0; mi < 2; mi++) {
#pragma unroll
    for (int ni = 0; ni < 4; ni++) {
      int col = n0 + nw + ni * 16 + (lane & 15);
      if (col < Ncol) {
        float bv = bias ? bias[col] : 0.f;
#pragma unroll
        for (int r = 0; r < 8; r++) {
          int row = m0 + mw + mi * 16 + r + ((lane >> 4) << 3);
          if (row < M) C[(size_t)row * Ncol + col] = acc[mi][ni][r] + bv;
        }
      }
    }
  }
}

// ---------------------------------------------------------------------------
// Edge attention kernels (wave per edge; lane handles 8 contiguous dims)
// q,k,v : [N, hc] with hc = HEADS*C ; logits/alpha : [E, 8]
// ---------------------------------------------------------------------------
__global__ __launch_bounds__(256)
void k_edge_logits(const float* __restrict__ q, const float* __restrict__ kk,
                   const float* __restrict__ vv, const int* __restrict__ ei,
                   float* __restrict__ logits, int hc, int C) {
  int wid = threadIdx.x >> 5, lane = threadIdx.x & 31;
  int e = blockIdx.x * 8 + wid;
  if (e >= NEDGES) return;
  int src = ei[e], dst = ei[NEDGES + e];
  if (lane == 0) __builtin_prefetch(&vv[(size_t)src * hc], 0, 1);  // warm L2 for scatter pass
  int d = lane * 8;
  float partial = 0.f;
  if (d < hc) {
    const f32x4* qp = reinterpret_cast<const f32x4*>(&q[(size_t)dst * hc + d]);
    const f32x4* kp = reinterpret_cast<const f32x4*>(&kk[(size_t)src * hc + d]);
    f32x4 q0 = qp[0], q1 = qp[1], k0 = kp[0], k1 = kp[1];
    partial = q0.x * k0.x + q0.y * k0.y + q0.z * k0.z + q0.w * k0.w +
              q1.x * k1.x + q1.y * k1.y + q1.z * k1.z + q1.w * k1.w;
  }
  int lph = C >> 3;  // lanes per head (4 for C=32, 1 for C=8)
  for (int off = 1; off < lph; off <<= 1) partial += __shfl_xor(partial, off, 32);
  if (d < hc && (lane & (lph - 1)) == 0) {
    int head = lane / lph;
    logits[(size_t)e * HEADS + head] = partial * rsqrtf((float)C);
  }
}

__global__ void k_seg_max(const float* __restrict__ logits, const int* __restrict__ ei,
                          float* __restrict__ m) {
  int idx = blockIdx.x * blockDim.x + threadIdx.x;
  if (idx >= NEDGES * HEADS) return;
  int e = idx >> 3, h = idx & 7;
  int dst = ei[NEDGES + e];
  atomicMaxF(&m[(size_t)dst * HEADS + h], logits[idx]);
}

__global__ void k_edge_exp(float* __restrict__ logits, const int* __restrict__ ei,
                           const float* __restrict__ m, float* __restrict__ s) {
  int idx = blockIdx.x * blockDim.x + threadIdx.x;
  if (idx >= NEDGES * HEADS) return;
  int e = idx >> 3, h = idx & 7;
  int dst = ei[NEDGES + e];
  float ee = __expf(logits[idx] - m[(size_t)dst * HEADS + h]);
  logits[idx] = ee;
  atomicAdd(&s[(size_t)dst * HEADS + h], ee);
}

__global__ __launch_bounds__(256)
void k_scatter(const float* __restrict__ vv, const float* __restrict__ alpha,
               const int* __restrict__ ei, const float* __restrict__ s,
               float* __restrict__ out, int hc, int C) {
  int wid = threadIdx.x >> 5, lane = threadIdx.x & 31;
  int e = blockIdx.x * 8 + wid;
  if (e >= NEDGES) return;
  int src = ei[e], dst = ei[NEDGES + e];
  int d = lane * 8;
  if (d >= hc) return;
  int lph = C >> 3;
  int head = lane / lph;
  float a = alpha[(size_t)e * HEADS + head];
  float coef = a / (s[(size_t)dst * HEADS + head] + 1e-16f);
  const f32x4* vp = reinterpret_cast<const f32x4*>(&vv[(size_t)src * hc + d]);
  f32x4 v0 = vp[0], v1 = vp[1];
  float* op = &out[(size_t)dst * hc + d];
  atomicAdd(op + 0, v0.x * coef); atomicAdd(op + 1, v0.y * coef);
  atomicAdd(op + 2, v0.z * coef); atomicAdd(op + 3, v0.w * coef);
  atomicAdd(op + 4, v1.x * coef); atomicAdd(op + 5, v1.y * coef);
  atomicAdd(op + 6, v1.z * coef); atomicAdd(op + 7, v1.w * coef);
}

// layer-3 only: mean over heads, out [N,64] -> om [N,8]
__global__ void k_head_mean(const float* __restrict__ out, float* __restrict__ om) {
  int idx = blockIdx.x * blockDim.x + threadIdx.x;
  if (idx >= NNODES * 8) return;
  int n = idx >> 3, c = idx & 7;
  float sum = 0.f;
#pragma unroll
  for (int h = 0; h < 8; h++) sum += out[(size_t)n * 64 + h * 8 + c];
  om[idx] = sum * 0.125f;
}

// beta-gated skip: dst = beta*xr + (1-beta)*out ; beta = sigmoid([out,xr,out-xr]@Wb)
__global__ __launch_bounds__(256)
void k_blend(const float* __restrict__ out, const float* __restrict__ xr,
             const float* __restrict__ Wb, float* __restrict__ dst, int S) {
  int wid = threadIdx.x >> 5, lane = threadIdx.x & 31;
  int n = blockIdx.x * 8 + wid;
  if (n >= NNODES) return;
  float z = 0.f;
  for (int j = lane; j < S; j += 32) {
    float o = out[(size_t)n * S + j];
    float r = xr[(size_t)n * S + j];
    z += o * Wb[j] + r * Wb[S + j] + (o - r) * Wb[2 * S + j];
  }
  z = wave_reduce_sum(z);
  float beta = 1.f / (1.f + __expf(-z));
  for (int j = lane; j < S; j += 32) {
    float o = out[(size_t)n * S + j];
    float r = xr[(size_t)n * S + j];
    dst[(size_t)n * S + j] = beta * r + (1.f - beta) * o;
  }
}

// layernorm(256) + relu, writes f32 in place and bf16 copy for next layer's GEMMs
__global__ __launch_bounds__(256)
void k_ln_relu_cast(float* __restrict__ h, const float* __restrict__ g,
                    const float* __restrict__ b, bf16* __restrict__ xb) {
  int wid = threadIdx.x >> 5, lane = threadIdx.x & 31;
  int n = blockIdx.x * 8 + wid;
  if (n >= NNODES) return;
  float* hp = &h[(size_t)n * 256 + lane * 8];
  union { f32x4 f4[2]; float f[8]; } u;
  u.f4[0] = *reinterpret_cast<const f32x4*>(hp);
  u.f4[1] = *reinterpret_cast<const f32x4*>(hp + 4);
  float sum = 0.f;
#pragma unroll
  for (int t = 0; t < 8; t++) sum += u.f[t];
  float mu = wave_reduce_sum(sum) * (1.f / 256.f);
  float vs = 0.f;
#pragma unroll
  for (int t = 0; t < 8; t++) { float dd = u.f[t] - mu; vs += dd * dd; }
  float var = wave_reduce_sum(vs) * (1.f / 256.f);
  float rstd = rsqrtf(var + 1e-5f);
  int dbase = lane * 8;
#pragma unroll
  for (int t = 0; t < 8; t++) {
    float y = (u.f[t] - mu) * rstd * g[dbase + t] + b[dbase + t];
    y = fmaxf(y, 0.f);
    hp[t] = y;
    xb[(size_t)n * 256 + dbase + t] = (bf16)y;
  }
}

// ---------------------------------------------------------------------------
// Host orchestration
// ---------------------------------------------------------------------------
extern "C" void kernel_launch(void* const* d_in, const int* in_sizes, int n_in,
                              void* d_out, int out_size, void* d_ws, size_t ws_size,
                              hipStream_t stream) {
  (void)in_sizes; (void)n_in; (void)out_size; (void)ws_size;

  const float* x  = (const float*)d_in[0];
  const int*   ei = (const int*)d_in[1];
  const float *Wq[3], *bq[3], *Wk[3], *bk[3], *Wv[3], *bv[3], *Wss[3], *bss[3], *Wb[3];
  for (int L = 0; L < 3; L++) {
    const int base = 2 + L * 9;
    Wq[L]  = (const float*)d_in[base + 0]; bq[L]  = (const float*)d_in[base + 1];
    Wk[L]  = (const float*)d_in[base + 2]; bk[L]  = (const float*)d_in[base + 3];
    Wv[L]  = (const float*)d_in[base + 4]; bv[L]  = (const float*)d_in[base + 5];
    Wss[L] = (const float*)d_in[base + 6]; bss[L] = (const float*)d_in[base + 7];
    Wb[L]  = (const float*)d_in[base + 8];
  }
  const float* ln_g[2] = {(const float*)d_in[29], (const float*)d_in[31]};
  const float* ln_b[2] = {(const float*)d_in[30], (const float*)d_in[32]};

  char* ws = (char*)d_ws;
  size_t off = 0;
  auto alloc = [&](size_t bytes) -> void* {
    void* p = ws + off;
    off += (bytes + 255) & ~(size_t)255;
    return p;
  };
  bf16*  xb     = (bf16*)alloc((size_t)NNODES * 256 * sizeof(bf16));
  bf16*  wqb    = (bf16*)alloc(256 * 256 * sizeof(bf16));
  bf16*  wkb    = (bf16*)alloc(256 * 256 * sizeof(bf16));
  bf16*  wvb    = (bf16*)alloc(256 * 256 * sizeof(bf16));
  bf16*  wsb    = (bf16*)alloc(256 * 256 * sizeof(bf16));
  float* qb     = (float*)alloc((size_t)NNODES * 256 * sizeof(float));
  float* kb     = (float*)alloc((size_t)NNODES * 256 * sizeof(float));
  float* vb     = (float*)alloc((size_t)NNODES * 256 * sizeof(float));
  float* xrb    = (float*)alloc((size_t)NNODES * 256 * sizeof(float));
  float* outb   = (float*)alloc((size_t)NNODES * 256 * sizeof(float));
  float* hbuf   = (float*)alloc((size_t)NNODES * 256 * sizeof(float));
  float* logits = (float*)alloc((size_t)NEDGES * 8 * sizeof(float));
  float* mbuf   = (float*)alloc((size_t)NNODES * 8 * sizeof(float));
  float* sbuf   = (float*)alloc((size_t)NNODES * 8 * sizeof(float));
  float* om     = (float*)alloc((size_t)NNODES * 8 * sizeof(float));

  const float NEG_INF = -__builtin_huge_valf();
  const int hcs[3]   = {256, 256, 64};
  const int skips[3] = {256, 256, 8};

  { // x -> bf16
    int n = NNODES * 256;
    k_cast_bf16<<<(n + 255) / 256, 256, 0, stream>>>(x, xb, n);
  }

  for (int L = 0; L < 3; L++) {
    int hc = hcs[L], S = skips[L], C = hc / 8;

    k_cast_bf16<<<(256 * hc + 255) / 256, 256, 0, stream>>>(Wq[L], wqb, 256 * hc);
    k_cast_bf16<<<(256 * hc + 255) / 256, 256, 0, stream>>>(Wk[L], wkb, 256 * hc);
    k_cast_bf16<<<(256 * hc + 255) / 256, 256, 0, stream>>>(Wv[L], wvb, 256 * hc);
    k_cast_bf16<<<(256 * S + 255) / 256, 256, 0, stream>>>(Wss[L], wsb, 256 * S);

    dim3 gg((NNODES + 127) / 128, (hc + 127) / 128);
    k_gemm_bf16<<<gg, 256, 0, stream>>>(xb, wqb, bq[L], qb, NNODES, hc);
    k_gemm_bf16<<<gg, 256, 0, stream>>>(xb, wkb, bk[L], kb, NNODES, hc);
    k_gemm_bf16<<<gg, 256, 0, stream>>>(xb, wvb, bv[L], vb, NNODES, hc);
    dim3 gs((NNODES + 127) / 128, (S + 127) / 128);
    k_gemm_bf16<<<gs, 256, 0, stream>>>(xb, wsb, bss[L], xrb, NNODES, S);

    k_fill<<<(NNODES * 8 + 255) / 256, 256, 0, stream>>>(mbuf, NEG_INF, NNODES * 8);
    k_fill<<<(NNODES * 8 + 255) / 256, 256, 0, stream>>>(sbuf, 0.f, NNODES * 8);
    k_fill<<<(NNODES * hc + 255) / 256, 256, 0, stream>>>(outb, 0.f, NNODES * hc);

    int eb = (NEDGES + 7) / 8;
    k_edge_logits<<<eb, 256, 0, stream>>>(qb, kb, vb, ei, logits, hc, C);
    k_seg_max<<<(NEDGES * 8 + 255) / 256, 256, 0, stream>>>(logits, ei, mbuf);
    k_edge_exp<<<(NEDGES * 8 + 255) / 256, 256, 0, stream>>>(logits, ei, mbuf, sbuf);
    k_scatter<<<eb, 256, 0, stream>>>(vb, logits, ei, sbuf, outb, hc, C);

    int nb = (NNODES + 7) / 8;
    if (L < 2) {
      k_blend<<<nb, 256, 0, stream>>>(outb, xrb, Wb[L], hbuf, 256);
      k_ln_relu_cast<<<nb, 256, 0, stream>>>(hbuf, ln_g[L], ln_b[L], xb);
    } else {
      k_head_mean<<<(NNODES * 8 + 255) / 256, 256, 0, stream>>>(outb, om);
      k_blend<<<nb, 256, 0, stream>>>(om, xrb, Wb[L], (float*)d_out, 8);
    }
  }
}
